// GAThlayer_90529320665246
// MI455X (gfx1250) — compile-verified, matrix operand order checked by reading, not run on hardware
//
#include <hip/hip_runtime.h>
#include <hip/hip_bf16.h>

// GAT layer for MI455X (gfx1250): f16 WMMA GEMM + 3-pass edge softmax + atomics + wave32 LayerNorm.

typedef __attribute__((ext_vector_type(16))) _Float16 v16h;
typedef __attribute__((ext_vector_type(8)))  _Float16 v8h;
typedef __attribute__((ext_vector_type(8)))  float    v8f;

#define IN_F   128
#define OUT_F  128
#define HEADS  4
#define HC     32
#define NEG_SLOPE 0.2f
#define LN_EPS 1e-5f
#define SM_EPS 1e-16f
#define LDSPAD 8          // +8 halves (16B) per row: keeps 16B alignment, breaks bank conflicts
#define ROWH  (IN_F + LDSPAD)

union frag16 { v16h v; v8h h[2]; };

// ---------------------------------------------------------------- init
__global__ __launch_bounds__(256)
void gat_init_kernel(float* __restrict__ agg, float* __restrict__ emax,
                     float* __restrict__ denom, int n) {
    int i = blockIdx.x * 256 + threadIdx.x;
    if (i < n * OUT_F) agg[i] = 0.0f;
    if (i < n * HEADS) { emax[i] = -INFINITY; denom[i] = 0.0f; }
}

// ---------------------------------------------------------------- GEMM h = x @ W  (WMMA f32_16x16x32_f16)
__global__ __launch_bounds__(128)
void gat_gemm_kernel(const float* __restrict__ x, const float* __restrict__ W,
                     float* __restrict__ h, int n) {
    __shared__ _Float16 sW[OUT_F][ROWH];   // W transposed: sW[col][k]
    __shared__ _Float16 sX[4][16][ROWH];   // per-wave 16-row strip of x

    const int tid  = threadIdx.x;
    const int wave = tid >> 5;
    const int lane = tid & 31;

    // stage W (128x128 f32 -> f16, transposed)
    for (int i = tid; i < IN_F * OUT_F; i += 128) {
        int k = i >> 7;          // row of W (k index)
        int c = i & 127;         // col of W (output index)
        sW[c][k] = (_Float16)W[i];
    }
    // stage this wave's 16 rows of x
    const int row0 = blockIdx.x * 64 + wave * 16;
    for (int i = lane; i < 16 * IN_F; i += 32) {
        int r = i >> 7;
        int k = i & 127;
        int node = row0 + r;
        sX[wave][r][k] = (node < n) ? (_Float16)x[(size_t)node * IN_F + k] : (_Float16)0.0f;
    }
    __syncthreads();

    const int mhalf = lane >> 4;    // k-group select (ISA 16-bit A/B layout)
    const int mrow  = lane & 15;    // row (A) / col (B) within tile

    v8f acc[8];
    for (int t = 0; t < 8; ++t)
        for (int j = 0; j < 8; ++j) acc[t][j] = 0.0f;

    for (int ks = 0; ks < 4; ++ks) {
        const int kb = ks * 32;
        frag16 a;
        a.h[0] = *(const v8h*)&sX[wave][mrow][kb + mhalf * 8];
        a.h[1] = *(const v8h*)&sX[wave][mrow][kb + 16 + mhalf * 8];
        for (int nt = 0; nt < 8; ++nt) {
            frag16 b;
            b.h[0] = *(const v8h*)&sW[nt * 16 + mrow][kb + mhalf * 8];
            b.h[1] = *(const v8h*)&sW[nt * 16 + mrow][kb + 16 + mhalf * 8];
            acc[nt] = __builtin_amdgcn_wmma_f32_16x16x32_f16(
                false, a.v, false, b.v, (short)0, acc[nt], false, false);
        }
    }

    // D layout: VGPR j -> M = (lane/16)*8 + j, N = lane%16
    for (int nt = 0; nt < 8; ++nt)
        for (int j = 0; j < 8; ++j) {
            int node = row0 + mhalf * 8 + j;
            if (node < n) h[(size_t)node * OUT_F + nt * 16 + mrow] = acc[nt][j];
        }
}

// ---------------------------------------------------------------- attention logits per node/head
__global__ __launch_bounds__(256)
void gat_alpha_kernel(const float* __restrict__ h, const float* __restrict__ a_src,
                      const float* __restrict__ a_dst, float* __restrict__ asrc,
                      float* __restrict__ adst, int n) {
    int i = blockIdx.x * 256 + threadIdx.x;      // i = node*HEADS + head
    if (i >= n * HEADS) return;
    int node = i >> 2, hd = i & 3;
    const float* hp = h + (size_t)node * OUT_F + hd * HC;
    const float* as = a_src + hd * HC;
    const float* ad = a_dst + hd * HC;
    float s = 0.0f, d = 0.0f;
    for (int c = 0; c < HC; ++c) { float v = hp[c]; s += v * as[c]; d += v * ad[c]; }
    asrc[i] = s;
    adst[i] = d;
}

__device__ inline void edge_nodes(const int* __restrict__ ei, int E, int e, int& s, int& d) {
    if (e < E) { s = ei[e]; d = ei[E + e]; }
    else       { s = d = e - E; }               // self loop
}

__device__ inline float leaky(float v) { return v > 0.0f ? v : NEG_SLOPE * v; }

__device__ inline void atomicMaxF32(float* addr, float val) {
    if (val >= 0.0f) atomicMax((int*)addr, __float_as_int(val));
    else             atomicMin((unsigned int*)addr, __float_as_uint(val));
}

// ---------------------------------------------------------------- pass 1: segment max
__global__ __launch_bounds__(256)
void gat_edge_max_kernel(const int* __restrict__ ei, const float* __restrict__ asrc,
                         const float* __restrict__ adst, float* __restrict__ emax,
                         int E, int tot) {
    int i = blockIdx.x * 256 + threadIdx.x;      // i = edge*HEADS + head
    if (i >= tot * HEADS) return;
    int e = i >> 2, hd = i & 3;
    int s, d; edge_nodes(ei, E, e, s, d);
    float v = leaky(asrc[(size_t)s * HEADS + hd] + adst[(size_t)d * HEADS + hd]);
    atomicMaxF32(&emax[(size_t)d * HEADS + hd], v);
}

// ---------------------------------------------------------------- pass 2: segment sum of exp
__global__ __launch_bounds__(256)
void gat_edge_expsum_kernel(const int* __restrict__ ei, const float* __restrict__ asrc,
                            const float* __restrict__ adst, const float* __restrict__ emax,
                            float* __restrict__ denom, int E, int tot) {
    int i = blockIdx.x * 256 + threadIdx.x;
    if (i >= tot * HEADS) return;
    int e = i >> 2, hd = i & 3;
    int s, d; edge_nodes(ei, E, e, s, d);
    size_t di = (size_t)d * HEADS + hd;
    float v = leaky(asrc[(size_t)s * HEADS + hd] + adst[di]);
    atomicAdd(&denom[di], expf(v - emax[di]));
}

// ---------------------------------------------------------------- pass 3: weighted scatter-aggregate
__global__ __launch_bounds__(256)
void gat_aggregate_kernel(const int* __restrict__ ei, const float* __restrict__ h,
                          const float* __restrict__ asrc, const float* __restrict__ adst,
                          const float* __restrict__ emax, const float* __restrict__ denom,
                          float* __restrict__ agg, int E, int tot) {
    long long gid = (long long)blockIdx.x * 256 + threadIdx.x;   // 32 lanes per edge
    int e    = (int)(gid >> 5);
    int lane = (int)(gid & 31);
    if (e >= tot) return;
    int s, d; edge_nodes(ei, E, e, s, d);
    int col0 = lane * 4;            // 4 consecutive cols, all in head col0/HC
    int hd   = lane >> 3;
    size_t di = (size_t)d * HEADS + hd;
    float a  = leaky(asrc[(size_t)s * HEADS + hd] + adst[di]);
    float al = expf(a - emax[di]) / (denom[di] + SM_EPS);
    const float4 hv = *(const float4*)(h + (size_t)s * OUT_F + col0);
    float* o = agg + (size_t)d * OUT_F + col0;
    atomicAdd(o + 0, al * hv.x);
    atomicAdd(o + 1, al * hv.y);
    atomicAdd(o + 2, al * hv.z);
    atomicAdd(o + 3, al * hv.w);
}

// ---------------------------------------------------------------- bias + LayerNorm + ReLU (wave32, in place)
__global__ __launch_bounds__(256)
void gat_ln_kernel(float* __restrict__ io, const float* __restrict__ bias,
                   const float* __restrict__ gamma, const float* __restrict__ beta, int n) {
    int wave = threadIdx.x >> 5;
    int lane = threadIdx.x & 31;
    int node = blockIdx.x * 8 + wave;
    if (node >= n) return;
    float* row = io + (size_t)node * OUT_F;
    float4 v = *(const float4*)(row + lane * 4);
    const float4 b  = *(const float4*)(bias  + lane * 4);
    v.x += b.x; v.y += b.y; v.z += b.z; v.w += b.w;
    float s  = v.x + v.y + v.z + v.w;
    float s2 = v.x * v.x + v.y * v.y + v.z * v.z + v.w * v.w;
    for (int off = 16; off > 0; off >>= 1) {
        s  += __shfl_xor(s,  off, 32);
        s2 += __shfl_xor(s2, off, 32);
    }
    float mu  = s * (1.0f / OUT_F);
    float var = s2 * (1.0f / OUT_F) - mu * mu;
    float inv = rsqrtf(var + LN_EPS);
    const float4 g  = *(const float4*)(gamma + lane * 4);
    const float4 be = *(const float4*)(beta  + lane * 4);
    float4 o;
    o.x = fmaxf((v.x - mu) * inv * g.x + be.x, 0.0f);
    o.y = fmaxf((v.y - mu) * inv * g.y + be.y, 0.0f);
    o.z = fmaxf((v.z - mu) * inv * g.z + be.z, 0.0f);
    o.w = fmaxf((v.w - mu) * inv * g.w + be.w, 0.0f);
    *(float4*)(row + lane * 4) = o;
}

// ---------------------------------------------------------------- launch
extern "C" void kernel_launch(void* const* d_in, const int* in_sizes, int n_in,
                              void* d_out, int out_size, void* d_ws, size_t ws_size,
                              hipStream_t stream) {
    const float* x     = (const float*)d_in[0];
    const int*   ei    = (const int*)  d_in[1];   // edge_index [2,E]
    const float* W     = (const float*)d_in[2];
    const float* a_src = (const float*)d_in[3];
    const float* a_dst = (const float*)d_in[4];
    const float* bias  = (const float*)d_in[5];
    const float* gamma = (const float*)d_in[6];
    const float* beta  = (const float*)d_in[7];

    const int n = in_sizes[0] / IN_F;
    const int E = in_sizes[1] / 2;
    const int tot = E + n;                         // edges + self loops

    float* agg   = (float*)d_out;                  // aggregation buffer == output (in-place LN)
    float* h     = (float*)d_ws;                   // [n, 128]
    float* asrc  = h    + (size_t)n * OUT_F;       // [n, H]
    float* adst  = asrc + (size_t)n * HEADS;
    float* emax  = adst + (size_t)n * HEADS;
    float* denom = emax + (size_t)n * HEADS;

    gat_init_kernel<<<(n * OUT_F + 255) / 256, 256, 0, stream>>>(agg, emax, denom, n);
    gat_gemm_kernel<<<(n + 63) / 64, 128, 0, stream>>>(x, W, h, n);
    gat_alpha_kernel<<<(n * HEADS + 255) / 256, 256, 0, stream>>>(h, a_src, a_dst, asrc, adst, n);
    gat_edge_max_kernel<<<(tot * HEADS + 255) / 256, 256, 0, stream>>>(ei, asrc, adst, emax, E, tot);
    gat_edge_expsum_kernel<<<(tot * HEADS + 255) / 256, 256, 0, stream>>>(ei, asrc, adst, emax, denom, E, tot);
    long long aggThreads = (long long)tot * 32;
    gat_aggregate_kernel<<<(unsigned)((aggThreads + 255) / 256), 256, 0, stream>>>(
        ei, h, asrc, adst, emax, denom, agg, E, tot);
    gat_ln_kernel<<<(n + 7) / 8, 256, 0, stream>>>(agg, bias, gamma, beta, n);
}
